// ResidualGCN_73735998537849
// MI455X (gfx1250) — matrix operand order repaired and checked by reading, not compiled, not run on hardware
//
#include <hip/hip_runtime.h>
#include <hip/hip_bf16.h>
#include <cstdint>
#include <cstddef>

typedef __attribute__((ext_vector_type(16))) _Float16 v16h;
typedef __attribute__((ext_vector_type(8)))  _Float16 v8h;
typedef __attribute__((ext_vector_type(8)))  float    v8f;

#define N_NODES 100000
#define N_EDGES 1600000
#define DIM     128
#define NLAB    64
#define BN_EPS  1e-5f

__device__ __forceinline__ void atomAddF(float* p, float v) {
  __hip_atomic_fetch_add(p, v, __ATOMIC_RELAXED, __HIP_MEMORY_SCOPE_AGENT);
}

// ---------------- graph normalization ----------------

__global__ void k_init_deg(float* deg) {
  int i = blockIdx.x * blockDim.x + threadIdx.x;
  if (i < N_NODES) deg[i] = 1.0f;            // self-loop weight
}

__global__ void k_deg_accum(const long long* __restrict__ col,
                            const float* __restrict__ w,
                            float* __restrict__ deg) {
  int e = blockIdx.x * blockDim.x + threadIdx.x;   // grid sized exactly E
  atomAddF(&deg[(size_t)col[e]], w[e]);
}

__global__ void k_dinv(float* __restrict__ dinv, float* __restrict__ snorm) {
  int i = blockIdx.x * blockDim.x + threadIdx.x;
  if (i < N_NODES) {
    float d = dinv[i];                 // holds degree
    float r = (d > 0.0f) ? rsqrtf(d) : 0.0f;
    dinv[i]  = r;
    snorm[i] = r * r;                  // self-loop norm (w=1)
  }
}

__global__ void k_norm(const long long* __restrict__ row,
                       const long long* __restrict__ col,
                       const float* __restrict__ w,
                       const float* __restrict__ dinv,
                       float* __restrict__ norm) {
  int e = blockIdx.x * blockDim.x + threadIdx.x;   // exactly E
  norm[e] = dinv[(size_t)row[e]] * w[e] * dinv[(size_t)col[e]];
}

// ---------------- dtype / layout prep ----------------

__global__ void k_f32_to_f16(const float* __restrict__ in, _Float16* __restrict__ out) {
  size_t i = (size_t)blockIdx.x * blockDim.x + threadIdx.x;   // exactly N*DIM
  out[i] = (_Float16)in[i];
}

// W is [128 x outw] k-major; produce Bt [outw x 128] n-major f16
__global__ void k_transpose_w(const float* __restrict__ W, _Float16* __restrict__ Bt, int outw) {
  int idx = blockIdx.x * blockDim.x + threadIdx.x;            // exactly outw*128
  int n = idx >> 7, k = idx & 127;
  Bt[idx] = (_Float16)W[(size_t)k * outw + n];
}

// ---------------- WMMA GEMM: out[M x OUTW] = A[M x 128] * Bt^T + bias ----------------

template <int OUTW, bool F16OUT>
__global__ void k_gemm_wmma(const _Float16* __restrict__ A,
                            const _Float16* __restrict__ Bt,
                            const float* __restrict__ bias,
                            void* __restrict__ outv) {
  constexpr int NT = OUTW / 16;      // n-tiles per row band
  constexpr int MT = 8 / NT;         // m-tiles per block (8 waves total)
  const int wave  = threadIdx.x >> 5;
  const int lane  = threadIdx.x & 31;
  const int ntile = wave % NT;
  const int mtile = wave / NT;
  const int m0    = blockIdx.x * (16 * MT) + mtile * 16;
  const int hi    = lane >> 4;
  const int lid   = lane & 15;

  const _Float16* arow = A  + (size_t)(m0 + lid) * DIM;         // A: m = lane&15
  const _Float16* brow = Bt + (size_t)(ntile * 16 + lid) * DIM; // B: n = lane&15

  v8f acc = {};
#pragma unroll
  for (int kb = 0; kb < 4; ++kb) {
    const int k0 = kb * 32;
    // A frag: k = k0 + hi*8 + {0..7}  and  +16 + {0..7}
    v8h a0 = *(const v8h*)(arow + k0 + hi * 8);
    v8h a1 = *(const v8h*)(arow + k0 + hi * 8 + 16);
    v16h a;
#pragma unroll
    for (int i = 0; i < 8; ++i) { a[i] = a0[i]; a[i + 8] = a1[i]; }
    // B frag: k = k0 + hi*8 + {0..7}  and  +16 + {0..7}, in n-major Bt
    v8h b0 = *(const v8h*)(brow + k0 + hi * 8);
    v8h b1 = *(const v8h*)(brow + k0 + hi * 8 + 16);
    v16h b;
#pragma unroll
    for (int i = 0; i < 8; ++i) { b[i] = b0[i]; b[i + 8] = b1[i]; }
    acc = __builtin_amdgcn_wmma_f32_16x16x32_f16(false, a, false, b,
                                                 (short)0, acc, false, false);
  }

  const int n = ntile * 16 + lid;
  const float bv = bias[n];
#pragma unroll
  for (int r = 0; r < 8; ++r) {
    const int m = m0 + hi * 8 + r;
    const float v = acc[r] + bv;
    if constexpr (F16OUT) ((_Float16*)outv)[(size_t)m * OUTW + n] = (_Float16)v;
    else                  ((float*)outv)[(size_t)m * OUTW + n]   = v;
  }
}

// ---------------- SpMM (scatter phase of GCN conv) ----------------

// agg[i,f] = snorm[i] * T[i,f]   (self-loop term; also initializes agg)
__global__ void k_selfloop(const float* __restrict__ snorm,
                           const _Float16* __restrict__ T,
                           float* __restrict__ agg) {
  size_t idx = (size_t)blockIdx.x * blockDim.x + threadIdx.x;   // exactly N*DIM
  agg[idx] = snorm[idx >> 7] * (float)T[idx];
}

// agg[col,f] += norm[e] * T[row,f]; 2 edges per 256-thread block
__global__ void k_edge_spmm(const long long* __restrict__ row,
                            const long long* __restrict__ col,
                            const float* __restrict__ norm,
                            const _Float16* __restrict__ T,
                            float* __restrict__ agg) {
  const int e = blockIdx.x * 2 + (threadIdx.x >> 7);
  const int f = threadIdx.x & 127;
  const size_t r = (size_t)row[e];
  const size_t c = (size_t)col[e];
  const float nv = norm[e];
  atomAddF(&agg[c * (size_t)DIM + f], nv * (float)T[r * (size_t)DIM + f]);
}

// ---------------- BatchNorm (training stats) + ReLU + residual ----------------

__global__ void k_zero_stats(float* stats) { stats[threadIdx.x] = 0.0f; } // 256 thr

__global__ void k_stats(const float* __restrict__ agg, float* __restrict__ stats) {
  const int f = threadIdx.x;     // 128 threads
  float s = 0.0f, s2 = 0.0f;
  for (int i = blockIdx.x; i < N_NODES; i += gridDim.x) {
    float v = agg[(size_t)i * DIM + f];
    s += v; s2 += v * v;
  }
  atomAddF(&stats[f], s);
  atomAddF(&stats[f + 128], s2);
}

// stats[256+f] = g*inv_std ; stats[384+f] = be - mu*g*inv_std
__global__ void k_finalize(float* __restrict__ stats,
                           const float* __restrict__ g,
                           const float* __restrict__ be) {
  const int f = threadIdx.x;     // 128 threads
  const float mu  = stats[f] / (float)N_NODES;
  const float var = stats[f + 128] / (float)N_NODES - mu * mu;
  const float a   = g[f] * rsqrtf(var + BN_EPS);
  stats[256 + f] = a;
  stats[384 + f] = be[f] - mu * a;
}

__global__ void k_bn_act(const float* __restrict__ agg,
                         const float* __restrict__ stats,
                         const _Float16* __restrict__ res,
                         _Float16* __restrict__ out,
                         int has_res) {
  size_t idx = (size_t)blockIdx.x * blockDim.x + threadIdx.x;   // exactly N*DIM
  const int f = (int)(idx & 127);
  float v = agg[idx] * stats[256 + f] + stats[384 + f];
  if (has_res) v += (float)res[idx];
  out[idx] = (_Float16)fmaxf(v, 0.0f);
}

// ---------------- launcher ----------------

extern "C" void kernel_launch(void* const* d_in, const int* in_sizes, int n_in,
                              void* d_out, int out_size, void* d_ws, size_t ws_size,
                              hipStream_t stream) {
  (void)in_sizes; (void)n_in; (void)out_size; (void)ws_size;

  const float*     x  = (const float*)d_in[0];
  const long long* ei = (const long long*)d_in[1];   // [2,E] int64
  const float*     ew = (const float*)d_in[2];
  const float* W1 = (const float*)d_in[3];  const float* b1 = (const float*)d_in[4];
  const float* g1 = (const float*)d_in[5];  const float* be1 = (const float*)d_in[6];
  const float* W2 = (const float*)d_in[7];  const float* b2 = (const float*)d_in[8];
  const float* g2 = (const float*)d_in[9];  const float* be2 = (const float*)d_in[10];
  const float* W3 = (const float*)d_in[11]; const float* b3 = (const float*)d_in[12];
  const float* g3 = (const float*)d_in[13]; const float* be3 = (const float*)d_in[14];
  const float* Wh = (const float*)d_in[15]; const float* bh = (const float*)d_in[16];
  float* out = (float*)d_out;

  const long long* e_row = ei;            // source
  const long long* e_col = ei + N_EDGES;  // target

  // workspace carve-up (256B aligned)
  char* ws = (char*)d_ws;
  size_t off = 0;
  auto carve = [&](size_t bytes) -> void* {
    void* p = ws + off;
    off = (off + bytes + 255) & ~(size_t)255;
    return p;
  };
  float*     dinv  = (float*)carve((size_t)N_NODES * 4);
  float*     snorm = (float*)carve((size_t)N_NODES * 4);
  float*     norm  = (float*)carve((size_t)N_EDGES * 4);
  float*     stats = (float*)carve(512 * 4);
  _Float16*  Bt    = (_Float16*)carve((size_t)DIM * DIM * 2);
  _Float16*  bufX  = (_Float16*)carve((size_t)N_NODES * DIM * 2);
  _Float16*  bufA  = (_Float16*)carve((size_t)N_NODES * DIM * 2);
  _Float16*  bufT  = (_Float16*)carve((size_t)N_NODES * DIM * 2);
  float*     agg   = (float*)carve((size_t)N_NODES * DIM * 4);

  const int GB_N    = (N_NODES + 255) / 256;      // 391
  const int GB_E    = N_EDGES / 256;              // 6250
  const int GB_ND   = (N_NODES * DIM) / 256;      // 50000
  const int GB_E2   = N_EDGES / 2;                // 800000
  const int GB_GEMM = N_NODES / 16;               // 6250
  const int GB_HEAD = N_NODES / 32;               // 3125

  // symmetric normalization
  k_init_deg <<<GB_N, 256, 0, stream>>>(dinv);
  k_deg_accum<<<GB_E, 256, 0, stream>>>(e_col, ew, dinv);
  k_dinv     <<<GB_N, 256, 0, stream>>>(dinv, snorm);
  k_norm     <<<GB_E, 256, 0, stream>>>(e_row, e_col, ew, dinv, norm);

  // x -> f16
  k_f32_to_f16<<<GB_ND, 256, 0, stream>>>(x, bufX);

  auto run_layer = [&](const _Float16* hin, const float* W, const float* b,
                       const float* g, const float* be,
                       const _Float16* res, _Float16* hout) {
    k_transpose_w<<<(DIM * DIM) / 256, 256, 0, stream>>>(W, Bt, DIM);
    k_gemm_wmma<DIM, true><<<GB_GEMM, 256, 0, stream>>>(hin, Bt, b, (void*)bufT);
    k_selfloop <<<GB_ND, 256, 0, stream>>>(snorm, bufT, agg);
    k_edge_spmm<<<GB_E2, 256, 0, stream>>>(e_row, e_col, norm, bufT, agg);
    k_zero_stats<<<1, 256, 0, stream>>>(stats);
    k_stats    <<<512, 128, 0, stream>>>(agg, stats);
    k_finalize <<<1, 128, 0, stream>>>(stats, g, be);
    k_bn_act   <<<GB_ND, 256, 0, stream>>>(agg, stats, res ? res : bufT, hout,
                                           res ? 1 : 0);
  };

  run_layer(bufX, W1, b1, g1, be1, nullptr, bufA);   // h1 -> bufA
  run_layer(bufA, W2, b2, g2, be2, bufA,    bufX);   // h2 -> bufX (res = h1)
  run_layer(bufX, W3, b3, g3, be3, bufX,    bufA);   // h3 -> bufA (res = h2)

  // head: [N x 64] = h3 @ Wh + bh, fp32 out
  k_transpose_w<<<(DIM * NLAB) / 256, 256, 0, stream>>>(Wh, Bt, NLAB);
  k_gemm_wmma<NLAB, false><<<GB_HEAD, 256, 0, stream>>>(bufA, Bt, bh, (void*)out);
}
